// TensorProductModel_18760417149587
// MI455X (gfx1250) — matrix operand order repaired and checked by reading, not compiled, not run on hardware
//
#include <hip/hip_runtime.h>

// ---------------------------------------------------------------------------
// DiffDock-style equivariant GNN forward for MI455X (gfx1250, wave32).
// Hot path: per-edge weight-generating MLPs as bf16 WMMA 16x16x32 GEMMs with
// fragment-shuffled weights staged into LDS via the Tensor Data Mover.
// ---------------------------------------------------------------------------

typedef __attribute__((ext_vector_type(16))) __bf16 v16bf;
typedef __attribute__((ext_vector_type(8)))  __bf16 v8bf;
typedef __attribute__((ext_vector_type(8)))  float  v8f;

#define NLIG   50000
#define NREC   50000
#define ELIG   300000
#define EREC   300000
#define ECROSS 300000
#define NB     8
#define HDIM   40   // max irreps feature width (16 + 12 + 12)

static __device__ __forceinline__ v8f v8zero() {
  v8f z;
#pragma unroll
  for (int i = 0; i < 8; ++i) z[i] = 0.0f;
  return z;
}

// A-operand fragment (ISA 16-bit A layout): lane m = l&15, K-half = l>>4;
// elements 0..7 -> K = kt*32 + half*8 + i ; elements 8..15 -> +16.
static __device__ __forceinline__ v16bf load_afrag(const __bf16* row, int kt, int half) {
  const int kb = kt * 32 + half * 8;
  v8bf lo = *(const v8bf*)(row + kb);
  v8bf hi = *(const v8bf*)(row + kb + 16);
  v16bf a;
#pragma unroll
  for (int i = 0; i < 8; ++i) { a[i] = lo[i]; a[8 + i] = hi[i]; }
  return a;
}

// ---------------------------------------------------------------------------
// Tensor Data Mover staging of weight fragments into LDS (gfx1250 TDM).
// 1-D descriptor: count=1, data_size=4B, tile = whole buffer.
// ---------------------------------------------------------------------------
#if __has_builtin(__builtin_amdgcn_tensor_load_to_lds) && \
    __has_builtin(__builtin_amdgcn_s_wait_tensorcnt)
#define HAVE_TDM 1
typedef __attribute__((ext_vector_type(4))) unsigned int u32x4;
typedef __attribute__((ext_vector_type(4))) int i32x4;
typedef __attribute__((ext_vector_type(8))) int i32x8;

static __device__ __forceinline__ void tdm_load_1d(const void* gsrc, void* ldsdst,
                                                   unsigned int dwords) {
  unsigned long long ga = (unsigned long long)gsrc;
  unsigned int la = (unsigned int)(unsigned long long)ldsdst;  // LDS addr = flat[31:0]
  u32x4 g0;
  g0[0] = 1u;                                                  // count = 1 (valid D#)
  g0[1] = la;                                                  // lds_addr
  g0[2] = (unsigned int)ga;                                    // global_addr[31:0]
  g0[3] = (unsigned int)((ga >> 32) & 0x01FFFFFFu) | (2u << 30);  // [56:32] | type=2
  i32x8 g1;
  g1[0] = 2 << 16;                                             // data_size = 4 bytes
  g1[1] = (int)((dwords & 0xFFFFu) << 16);                     // tensor_dim0[15:0]
  g1[2] = (int)(dwords >> 16);                                 // tensor_dim0[31:16]
  g1[3] = (int)((dwords & 0xFFFFu) << 16);                     // tile_dim0
  g1[4] = 0;                                                   // tile_dim1/2 unused
  g1[5] = (int)dwords;                                         // tensor_dim0_stride lo
  g1[6] = 0;
  g1[7] = 0;
  i32x4 z4 = {0, 0, 0, 0};
#if defined(__clang_major__) && (__clang_major__ >= 23)
  i32x8 z8 = {0, 0, 0, 0, 0, 0, 0, 0};
  __builtin_amdgcn_tensor_load_to_lds(g0, g1, z4, z4, z8, 0);
#else
  __builtin_amdgcn_tensor_load_to_lds(g0, g1, z4, z4, 0);
#endif
}
#endif

// ---------------------------------------------------------------------------
// small glue kernels
// ---------------------------------------------------------------------------

__global__ void k_zero(float* __restrict__ p, long n) {
  long i = (long)blockIdx.x * blockDim.x + threadIdx.x;
  if (i < n) p[i] = 0.0f;
}

__global__ void k_sin_emb(const float* __restrict__ t, float* __restrict__ emb) {
  int b = blockIdx.x;
  int k = threadIdx.x;
  if (k >= 16) return;
  float fr = __expf(-9.210340371976184f * (float)k / 16.0f);  // ln(1e4)
  float a  = t[b] * 100.0f * fr;
  emb[b * 32 + k]      = __sinf(a);
  emb[b * 32 + 16 + k] = __cosf(a);
}

// h[n, :16] = concat(x16, emb[batch[n]]) @ W(48x16) + b ; h[n,16:40] = 0
__global__ void k_node_init(const float* __restrict__ x, const int* __restrict__ batch,
                            const float* __restrict__ emb,
                            const float* __restrict__ W, const float* __restrict__ bias,
                            float* __restrict__ h, int N) {
  int n = blockIdx.x * blockDim.x + threadIdx.x;
  if (n >= N) return;
  float in[48];
#pragma unroll
  for (int i = 0; i < 16; ++i) in[i] = x[(long)n * 16 + i];
  const float* e = emb + batch[n] * 32;
#pragma unroll
  for (int i = 0; i < 32; ++i) in[16 + i] = e[i];
  float out[16];
#pragma unroll
  for (int o = 0; o < 16; ++o) out[o] = bias[o];
  for (int k = 0; k < 48; ++k) {
    float v = in[k];
#pragma unroll
    for (int o = 0; o < 16; ++o) out[o] += v * W[k * 16 + o];
  }
  float* hp = h + (long)n * HDIM;
#pragma unroll
  for (int o = 0; o < 16; ++o) hp[o] = out[o];
#pragma unroll
  for (int o = 16; o < HDIM; ++o) hp[o] = 0.0f;
}

__global__ void k_edge_geom(const float* __restrict__ posS, const float* __restrict__ posD,
                            const int* __restrict__ src, const int* __restrict__ dst,
                            float* __restrict__ d, float* __restrict__ sh, int E) {
  int e = blockIdx.x * blockDim.x + threadIdx.x;
  if (e >= E) return;
  int s = src[e], t = dst[e];
  float vx = posD[(long)t * 3 + 0] - posS[(long)s * 3 + 0];
  float vy = posD[(long)t * 3 + 1] - posS[(long)s * 3 + 1];
  float vz = posD[(long)t * 3 + 2] - posS[(long)s * 3 + 2];
  float dd = sqrtf(vx * vx + vy * vy + vz * vz);
  float inv = 1.0f / fmaxf(dd, 1e-8f);
  const float S3 = 1.7320508075688772f;
  d[e] = dd;
  sh[(long)e * 4 + 0] = 1.0f;
  sh[(long)e * 4 + 1] = S3 * vx * inv;
  sh[(long)e * 4 + 2] = S3 * vy * inv;
  sh[(long)e * 4 + 3] = S3 * vz * inv;
}

// generic small edge-attribute MLP: in = [feat4?, emb, gauss] (order 0)
//                                   or [gauss, emb]           (order 1)
__global__ void k_edge_mlp(const float* __restrict__ feat4,
                           const int* __restrict__ nidx, const int* __restrict__ batch,
                           const float* __restrict__ emb, const float* __restrict__ d,
                           float gstop,
                           const float* __restrict__ W1, const float* __restrict__ b1,
                           const float* __restrict__ W2, const float* __restrict__ b2,
                           float* __restrict__ out, int E, int din, int order) {
  int e = blockIdx.x * blockDim.x + threadIdx.x;
  if (e >= E) return;
  float in[68];
  int f4 = feat4 ? 4 : 0;
  int go = (order == 0) ? (f4 + 32) : 0;
  int eo = (order == 0) ? f4 : 32;
  if (feat4) {
#pragma unroll
    for (int i = 0; i < 4; ++i) in[i] = feat4[(long)e * 4 + i];
  }
  int node = nidx ? nidx[e] : e;
  const float* em = emb + batch[node] * 32;
#pragma unroll
  for (int i = 0; i < 32; ++i) in[eo + i] = em[i];
  float step  = gstop / 31.0f;
  float coeff = -0.5f / (step * step);
  float dd = d[e];
#pragma unroll
  for (int k = 0; k < 32; ++k) {
    float tt = dd - step * (float)k;
    in[go + k] = __expf(coeff * tt * tt);
  }
  float h1[16];
#pragma unroll
  for (int o = 0; o < 16; ++o) h1[o] = b1[o];
  for (int k = 0; k < din; ++k) {
    float v = in[k];
#pragma unroll
    for (int o = 0; o < 16; ++o) h1[o] += v * W1[k * 16 + o];
  }
#pragma unroll
  for (int o = 0; o < 16; ++o) h1[o] = fmaxf(h1[o], 0.0f);
  float o2[16];
#pragma unroll
  for (int o = 0; o < 16; ++o) o2[o] = b2[o];
#pragma unroll
  for (int k = 0; k < 16; ++k) {
    float v = h1[k];
#pragma unroll
    for (int o = 0; o < 16; ++o) o2[o] += v * W2[k * 16 + o];
  }
#pragma unroll
  for (int o = 0; o < 16; ++o) out[(long)e * 16 + o] = o2[o];
}

// Build per-edge FC input, bf16, zero K-padded, 16-byte vector stores:
// X[e] = [ea(16), hA[ia[e],:16]?, hB[ib[e],:16]?] -> K0pad cols
__global__ void k_build_X(const float* __restrict__ ea,
                          const float* __restrict__ hA, const int* __restrict__ ia,
                          const float* __restrict__ hB, const int* __restrict__ ib,
                          __bf16* __restrict__ X, int E, int K0pad) {
  int e = blockIdx.x * blockDim.x + threadIdx.x;
  if (e >= E) return;
  __bf16 buf[64];
#pragma unroll
  for (int i = 0; i < 16; ++i) buf[i] = (__bf16)ea[(long)e * 16 + i];
  int p = 16;
  if (hA) {
    const float* a = hA + (long)(ia ? ia[e] : e) * HDIM;
#pragma unroll
    for (int i = 0; i < 16; ++i) buf[16 + i] = (__bf16)a[i];
    p = 32;
  }
  if (hB) {
    const float* b = hB + (long)(ib ? ib[e] : e) * HDIM;
#pragma unroll
    for (int i = 0; i < 16; ++i) buf[32 + i] = (__bf16)b[i];
    p = 48;
  }
  for (; p < K0pad; ++p) buf[p] = (__bf16)0.0f;
  __bf16* xp = X + (long)e * K0pad;
  for (int i = 0; i < K0pad; i += 8) *(v8bf*)(xp + i) = *(const v8bf*)(buf + i);
}

// Pack fp32 W[K,N] -> bf16 fragments in WMMA B-operand order:
// frag = nt*NKT + kt ; within frag: lane*16 + elem ;
// k = kt*32 + (lane>>4)*8 + elem (elem<8), kt*32 + 16 + (lane>>4)*8 + elem-8 (elem>=8) ; n = nt*16 + (lane&15) ; zero padded.
__global__ void k_pack_w_shuf(const float* __restrict__ W, __bf16* __restrict__ Wb,
                              int K, int N, int NKT, int NNT) {
  int i = blockIdx.x * blockDim.x + threadIdx.x;
  int total = NKT * NNT * 512;
  if (i >= total) return;
  int frag = i >> 9;
  int r    = i & 511;
  int lane = r >> 4;
  int el   = r & 15;
  int kt = frag % NKT;
  int nt = frag / NKT;
  int k = kt * 32 + (lane >> 4) * 8 + (el < 8 ? el : 8 + el);
  int n = nt * 16 + (lane & 15);
  float v = (k < K && n < N) ? W[k * N + n] : 0.0f;
  Wb[i] = (__bf16)v;
}

// ---------------------------------------------------------------------------
// WMMA edge-MLP:  Y[E,K2] = relu(X[E,K0p]@W1 + b1) @ W2 + b2   (bf16, f32 acc)
// Weights staged in LDS (TDM when available); one wave per 16-edge tile;
// hidden tile transposed D->A through LDS.
// ---------------------------------------------------------------------------
template <int K0P, int K1N, int K2N>
__global__ __launch_bounds__(256) void k_mlp_wmma(
    const __bf16* __restrict__ X, int E,
    const float* __restrict__ b1,
    const __bf16* __restrict__ W1g,   // shuffled fragments, FR1*512 bf16
    const __bf16* __restrict__ W2g,   // shuffled fragments, FR2*512 bf16
    const float* __restrict__ b2,
    float* __restrict__ Y) {
  constexpr int NK1 = K0P / 32;
  constexpr int NN1 = K1N / 16;
  constexpr int NK2 = (K1N > 32) ? 2 : 1;
  constexpr int NN2 = K2N / 16;
  constexpr int FR1 = NK1 * NN1;
  constexpr int FR2 = NK2 * NN2;
  __shared__ __bf16 w1s[FR1 * 512];
  __shared__ __bf16 w2s[FR2 * 512];
  __shared__ __bf16 lds_h[8][16][64];

  const int wave = threadIdx.x >> 5;
  const int lane = threadIdx.x & 31;
  const int half = lane >> 4;
  const int mloc = lane & 15;
  const long m0  = ((long)blockIdx.x * 8 + wave) * 16;

  // ---- stage shuffled weights into LDS ----------------------------------
#ifdef HAVE_TDM
  if (threadIdx.x == 0) {
    tdm_load_1d(W1g, w1s, (unsigned)(FR1 * 512 * 2 / 4));
    tdm_load_1d(W2g, w2s, (unsigned)(FR2 * 512 * 2 / 4));
    __builtin_amdgcn_s_wait_tensorcnt(0);
  }
#else
  for (int i = threadIdx.x * 16; i < FR1 * 512; i += 256 * 16)
    *(v16bf*)(w1s + i) = *(const v16bf*)(W1g + i);
  for (int i = threadIdx.x * 16; i < FR2 * 512; i += 256 * 16)
    *(v16bf*)(w2s + i) = *(const v16bf*)(W2g + i);
#endif

  // zero this wave's hidden slot (provides the K-pad for GEMM2)
  for (int i = lane; i < 16 * 64; i += 32) (&lds_h[wave][0][0])[i] = (__bf16)0.0f;
  __syncthreads();  // weights + zeroed hidden visible

  long arowi = m0 + mloc;
  if (arowi > (long)E - 1) arowi = E - 1;
  const __bf16* arow = X + arowi * K0P;

  // ---- GEMM1: hidden = relu(X @ W1 + b1) --------------------------------
#pragma unroll
  for (int nt = 0; nt < NN1; ++nt) {
    v8f c = v8zero();
#pragma unroll
    for (int kt = 0; kt < NK1; ++kt) {
      v16bf a = load_afrag(arow, kt, half);
      v16bf b = *(const v16bf*)(w1s + ((nt * NK1 + kt) << 9) + (lane << 4));
      c = __builtin_amdgcn_wmma_f32_16x16x32_bf16(false, a, false, b, (short)0, c,
                                                  false, false);
    }
    const int n = nt * 16 + mloc;
    const float bias = b1[n];
#pragma unroll
    for (int r = 0; r < 8; ++r) {          // D: elem r -> row (r + 8*half), col n
      float v = c[r] + bias;
      v = v > 0.0f ? v : 0.0f;
      lds_h[wave][r + 8 * half][n] = (__bf16)v;
    }
  }
  __syncthreads();  // hidden write -> read

  // ---- GEMM2: Y = hidden @ W2 + b2 --------------------------------------
  const __bf16* hrow = &lds_h[wave][mloc][0];
#pragma unroll
  for (int nt = 0; nt < NN2; ++nt) {
    v8f c = v8zero();
#pragma unroll
    for (int kt = 0; kt < NK2; ++kt) {
      v16bf a = load_afrag(hrow, kt, half);
      v16bf b = *(const v16bf*)(w2s + ((nt * NK2 + kt) << 9) + (lane << 4));
      c = __builtin_amdgcn_wmma_f32_16x16x32_bf16(false, a, false, b, (short)0, c,
                                                  false, false);
    }
    const int n = nt * 16 + mloc;
    const float bias = b2[n];
#pragma unroll
    for (int r = 0; r < 8; ++r) {
      long grow = m0 + r + 8 * half;
      if (grow < E) Y[grow * (long)K2N + n] = c[r] + bias;
    }
  }
}

// ---------------------------------------------------------------------------
// per-edge tensor product + atomic scatter-mean accumulation
// MODE 0: in=[16x0e]            out=[16x0e,4x1o]          wn=320
// MODE 1: in=[16x0e,4x1o]       out=[16x0e,4x1o,4x1e]     wn=416
// MODE 2: in=[16x0e,4x1o,4x1e]  out=[2x1o,2x1e]           wn=64
// ---------------------------------------------------------------------------
template <int MODE>
__global__ void k_tp(const float* __restrict__ Y, int K2,
                     const float* __restrict__ h, const int* __restrict__ gather,
                     const float* __restrict__ sh4, const int* __restrict__ scatter,
                     float* __restrict__ acc, float* __restrict__ cnt,
                     int E, int accw) {
  int e = blockIdx.x * blockDim.x + threadIdx.x;
  if (e >= E) return;
  const float* w = Y + (long)e * K2;
  const float* s = sh4 + (long)e * 4;
  float s0 = s[0], s1 = s[1], s2 = s[2], s3 = s[3];
  const float* x = h + (long)(gather ? gather[e] : e) * HDIM;
  float* ap = acc + (long)scatter[e] * accw;
  const float INV_S2 = 0.7071067811865476f;
  const float INV_S3 = 0.5773502691896258f;

  if (MODE == 0) {
    float t0[16], t1[4];
#pragma unroll
    for (int o = 0; o < 16; ++o) t0[o] = 0.0f;
#pragma unroll
    for (int o = 0; o < 4; ++o) t1[o] = 0.0f;
    for (int m = 0; m < 16; ++m) {
      float xv = x[m];
      const float* w0 = w + m * 16;
#pragma unroll
      for (int o = 0; o < 16; ++o) t0[o] += xv * w0[o];
      const float* w1 = w + 256 + m * 4;
#pragma unroll
      for (int o = 0; o < 4; ++o) t1[o] += xv * w1[o];
    }
    const float f = 0.25f;  // 1/sqrt(16)
#pragma unroll
    for (int o = 0; o < 16; ++o) atomicAdd(&ap[o], s0 * t0[o] * f);
#pragma unroll
    for (int o = 0; o < 4; ++o) {
      atomicAdd(&ap[16 + o * 3 + 0], t1[o] * s1 * f);
      atomicAdd(&ap[16 + o * 3 + 1], t1[o] * s2 * f);
      atomicAdd(&ap[16 + o * 3 + 2], t1[o] * s3 * f);
    }
  } else if (MODE == 1) {
    float x1[4][3];
#pragma unroll
    for (int m = 0; m < 4; ++m)
#pragma unroll
      for (int c = 0; c < 3; ++c) x1[m][c] = x[16 + m * 3 + c];
    float rp1[4];
#pragma unroll
    for (int m = 0; m < 4; ++m)
      rp1[m] = (x1[m][0] * s1 + x1[m][1] * s2 + x1[m][2] * s3) * INV_S3;
    float t0[16];
#pragma unroll
    for (int o = 0; o < 16; ++o) t0[o] = 0.0f;
    for (int m = 0; m < 16; ++m) {
      float xv = x[m] * s0;
      const float* w0 = w + m * 16;
#pragma unroll
      for (int o = 0; o < 16; ++o) t0[o] += xv * w0[o];
    }
#pragma unroll
    for (int m = 0; m < 4; ++m) {
      float rv = rp1[m];
      const float* w1 = w + 256 + m * 16;
#pragma unroll
      for (int o = 0; o < 16; ++o) t0[o] += rv * w1[o];
    }
    const float f0 = 0.22360679774997896f;  // 1/sqrt(20)
#pragma unroll
    for (int o = 0; o < 16; ++o) atomicAdd(&ap[o], t0[o] * f0);
    float t2[4];
#pragma unroll
    for (int o = 0; o < 4; ++o) t2[o] = 0.0f;
    for (int m = 0; m < 16; ++m) {
      float xv = x[m];
      const float* w2 = w + 320 + m * 4;
#pragma unroll
      for (int o = 0; o < 4; ++o) t2[o] += xv * w2[o];
    }
    float o1[4][3];
#pragma unroll
    for (int o = 0; o < 4; ++o) {
      o1[o][0] = t2[o] * s1; o1[o][1] = t2[o] * s2; o1[o][2] = t2[o] * s3;
    }
#pragma unroll
    for (int m = 0; m < 4; ++m) {
      const float* w3 = w + 384 + m * 4;
#pragma unroll
      for (int o = 0; o < 4; ++o) {
        float wv = w3[o] * s0;
        o1[o][0] += x1[m][0] * wv; o1[o][1] += x1[m][1] * wv; o1[o][2] += x1[m][2] * wv;
      }
    }
#pragma unroll
    for (int o = 0; o < 4; ++o)
#pragma unroll
      for (int c = 0; c < 3; ++c) atomicAdd(&ap[16 + o * 3 + c], o1[o][c] * f0);
    float cr[4][3];
#pragma unroll
    for (int m = 0; m < 4; ++m) {
      cr[m][0] = (x1[m][1] * s3 - x1[m][2] * s2) * INV_S2;
      cr[m][1] = (x1[m][2] * s1 - x1[m][0] * s3) * INV_S2;
      cr[m][2] = (x1[m][0] * s2 - x1[m][1] * s1) * INV_S2;
    }
    float o2[4][3];
#pragma unroll
    for (int o = 0; o < 4; ++o)
#pragma unroll
      for (int c = 0; c < 3; ++c) o2[o][c] = 0.0f;
#pragma unroll
    for (int m = 0; m < 4; ++m) {
      const float* w4 = w + 400 + m * 4;
#pragma unroll
      for (int o = 0; o < 4; ++o) {
        float wv = w4[o];
#pragma unroll
        for (int c = 0; c < 3; ++c) o2[o][c] += cr[m][c] * wv;
      }
    }
#pragma unroll
    for (int o = 0; o < 4; ++o)
#pragma unroll
      for (int c = 0; c < 3; ++c) atomicAdd(&ap[28 + o * 3 + c], o2[o][c] * 0.5f);
  } else {
    float x1[4][3], x2[4][3];
#pragma unroll
    for (int m = 0; m < 4; ++m)
#pragma unroll
      for (int c = 0; c < 3; ++c) {
        x1[m][c] = x[16 + m * 3 + c];
        x2[m][c] = x[28 + m * 3 + c];
      }
    float t0[2] = {0.0f, 0.0f};
    for (int m = 0; m < 16; ++m) {
      float xv = x[m];
      t0[0] += xv * w[m * 2 + 0];
      t0[1] += xv * w[m * 2 + 1];
    }
    float o0[2][3];
#pragma unroll
    for (int o = 0; o < 2; ++o) {
      o0[o][0] = t0[o] * s1; o0[o][1] = t0[o] * s2; o0[o][2] = t0[o] * s3;
    }
#pragma unroll
    for (int m = 0; m < 4; ++m)
#pragma unroll
      for (int o = 0; o < 2; ++o) {
        float wv = w[32 + m * 2 + o] * s0;
#pragma unroll
        for (int c = 0; c < 3; ++c) o0[o][c] += x1[m][c] * wv;
      }
#pragma unroll
    for (int m = 0; m < 4; ++m) {
      float c0 = (x2[m][1] * s3 - x2[m][2] * s2) * INV_S2;
      float c1 = (x2[m][2] * s1 - x2[m][0] * s3) * INV_S2;
      float c2 = (x2[m][0] * s2 - x2[m][1] * s1) * INV_S2;
#pragma unroll
      for (int o = 0; o < 2; ++o) {
        float wv = w[40 + m * 2 + o];
        o0[o][0] += c0 * wv; o0[o][1] += c1 * wv; o0[o][2] += c2 * wv;
      }
    }
    const float f0 = 0.20412414523193154f;  // 1/sqrt(24)
#pragma unroll
    for (int o = 0; o < 2; ++o)
#pragma unroll
      for (int c = 0; c < 3; ++c) atomicAdd(&ap[o * 3 + c], o0[o][c] * f0);
    float o1[2][3];
#pragma unroll
    for (int o = 0; o < 2; ++o)
#pragma unroll
      for (int c = 0; c < 3; ++c) o1[o][c] = 0.0f;
#pragma unroll
    for (int m = 0; m < 4; ++m) {
      float c0 = (x1[m][1] * s3 - x1[m][2] * s2) * INV_S2;
      float c1 = (x1[m][2] * s1 - x1[m][0] * s3) * INV_S2;
      float c2 = (x1[m][0] * s2 - x1[m][1] * s1) * INV_S2;
#pragma unroll
      for (int o = 0; o < 2; ++o) {
        float wv = w[48 + m * 2 + o];
        o1[o][0] += c0 * wv; o1[o][1] += c1 * wv; o1[o][2] += c2 * wv;
      }
    }
#pragma unroll
    for (int m = 0; m < 4; ++m)
#pragma unroll
      for (int o = 0; o < 2; ++o) {
        float wv = w[56 + m * 2 + o] * s0;
#pragma unroll
        for (int c = 0; c < 3; ++c) o1[o][c] += x2[m][c] * wv;
      }
    const float f1 = 0.35355339059327373f;  // 1/sqrt(8)
#pragma unroll
    for (int o = 0; o < 2; ++o)
#pragma unroll
      for (int c = 0; c < 3; ++c) atomicAdd(&ap[6 + o * 3 + c], o1[o][c] * f1);
  }
  atomicAdd(&cnt[scatter[e]], 1.0f);
}

// h = pad(h) + acc1/max(c1,1) + acc2/max(c2,1)
__global__ void k_node_update(float* __restrict__ h,
                              const float* __restrict__ a1, const float* __restrict__ c1,
                              const float* __restrict__ a2, const float* __restrict__ c2,
                              int N, int newdim) {
  int n = blockIdx.x * blockDim.x + threadIdx.x;
  if (n >= N) return;
  float i1 = 1.0f / fmaxf(c1[n], 1.0f);
  float i2 = 1.0f / fmaxf(c2[n], 1.0f);
  float* hp = h + (long)n * HDIM;
  const float* p1 = a1 + (long)n * HDIM;
  const float* p2 = a2 + (long)n * HDIM;
  for (int j = 0; j < newdim; ++j) hp[j] = hp[j] + p1[j] * i1 + p2[j] * i2;
  for (int j = newdim; j < HDIM; ++j) hp[j] = 0.0f;
}

__global__ void k_center_acc(const float* __restrict__ pos, const int* __restrict__ batch,
                             float* __restrict__ centers, float* __restrict__ cnt, int N) {
  int n = blockIdx.x * blockDim.x + threadIdx.x;
  if (n >= N) return;
  int b = batch[n];
  atomicAdd(&centers[b * 3 + 0], pos[(long)n * 3 + 0]);
  atomicAdd(&centers[b * 3 + 1], pos[(long)n * 3 + 1]);
  atomicAdd(&centers[b * 3 + 2], pos[(long)n * 3 + 2]);
  atomicAdd(&cnt[b], 1.0f);
}

__global__ void k_center_div(float* __restrict__ centers, const float* __restrict__ cnt) {
  int b = threadIdx.x;
  if (b >= NB) return;
  float c = 1.0f / fmaxf(cnt[b], 1.0f);
  centers[b * 3 + 0] *= c; centers[b * 3 + 1] *= c; centers[b * 3 + 2] *= c;
}

__global__ void k_cen_geom(const float* __restrict__ pos, const int* __restrict__ batch,
                           const float* __restrict__ centers,
                           float* __restrict__ d, float* __restrict__ sh, int N) {
  int n = blockIdx.x * blockDim.x + threadIdx.x;
  if (n >= N) return;
  const float* c = centers + batch[n] * 3;
  float vx = pos[(long)n * 3 + 0] - c[0];
  float vy = pos[(long)n * 3 + 1] - c[1];
  float vz = pos[(long)n * 3 + 2] - c[2];
  float dd = sqrtf(vx * vx + vy * vy + vz * vz);
  float inv = 1.0f / fmaxf(dd, 1e-8f);
  const float S3 = 1.7320508075688772f;
  d[n] = dd;
  sh[(long)n * 4 + 0] = 1.0f;
  sh[(long)n * 4 + 1] = S3 * vx * inv;
  sh[(long)n * 4 + 2] = S3 * vy * inv;
  sh[(long)n * 4 + 3] = S3 * vz * inv;
}

__global__ void k_final(const float* __restrict__ gp, const float* __restrict__ gcnt,
                        const float* __restrict__ emb,
                        const float* __restrict__ tW1, const float* __restrict__ tb1,
                        const float* __restrict__ tW2, const float* __restrict__ tb2,
                        const float* __restrict__ rW1, const float* __restrict__ rb1,
                        const float* __restrict__ rW2, const float* __restrict__ rb2,
                        float* __restrict__ out) {
  int b = threadIdx.x;
  if (b >= NB) return;
  float ic = 1.0f / fmaxf(gcnt[b], 1.0f);
  float g[12];
#pragma unroll
  for (int i = 0; i < 12; ++i) g[i] = gp[b * 12 + i] * ic;
  float tr[3] = {g[0] + g[6], g[1] + g[7], g[2] + g[8]};
  float ro[3] = {g[3] + g[9], g[4] + g[10], g[5] + g[11]};
  float trn = fmaxf(sqrtf(tr[0] * tr[0] + tr[1] * tr[1] + tr[2] * tr[2]), 1e-6f);
  float ron = fmaxf(sqrtf(ro[0] * ro[0] + ro[1] * ro[1] + ro[2] * ro[2]), 1e-6f);
  float in[33], h1[16];
  in[0] = trn;
#pragma unroll
  for (int i = 0; i < 32; ++i) in[1 + i] = emb[b * 32 + i];
#pragma unroll
  for (int o = 0; o < 16; ++o) h1[o] = tb1[o];
  for (int k = 0; k < 33; ++k) {
    float v = in[k];
#pragma unroll
    for (int o = 0; o < 16; ++o) h1[o] += v * tW1[k * 16 + o];
  }
  float st = tb2[0];
#pragma unroll
  for (int k = 0; k < 16; ++k) st += fmaxf(h1[k], 0.0f) * tW2[k];
#pragma unroll
  for (int c = 0; c < 3; ++c) out[b * 6 + c] = tr[c] / trn * st;
  in[0] = ron;
#pragma unroll
  for (int o = 0; o < 16; ++o) h1[o] = rb1[o];
  for (int k = 0; k < 33; ++k) {
    float v = in[k];
#pragma unroll
    for (int o = 0; o < 16; ++o) h1[o] += v * rW1[k * 16 + o];
  }
  float sr = rb2[0];
#pragma unroll
  for (int k = 0; k < 16; ++k) sr += fmaxf(h1[k], 0.0f) * rW2[k];
#pragma unroll
  for (int c = 0; c < 3; ++c) out[b * 6 + 3 + c] = ro[c] / ron * sr;
}

// ---------------------------------------------------------------------------
// host orchestration
// ---------------------------------------------------------------------------
extern "C" void kernel_launch(void* const* d_in, const int* in_sizes, int n_in,
                              void* d_out, int out_size, void* d_ws, size_t ws_size,
                              hipStream_t stream) {
  (void)in_sizes; (void)n_in; (void)out_size; (void)ws_size;
  auto P = [&](int i) { return (const float*)d_in[i]; };

  const float* lig_x   = P(0);
  const float* rec_x   = P(1);
  const float* lig_pos = P(2);
  const float* rec_pos = P(3);
  const float* lig_ef  = P(4);
  const float* t_tr    = P(5);
  // params flattened in dict-insertion order, indices 6..69
  const int* lig_ei = (const int*)d_in[70];
  const int* rec_ei = (const int*)d_in[71];
  const int* crs_ei = (const int*)d_in[72];
  const int* lig_batch = (const int*)d_in[73];
  const int* rec_batch = (const int*)d_in[74];
  const int *lsrc = lig_ei, *ldst = lig_ei + ELIG;
  const int *rsrc = rec_ei, *rdst = rec_ei + EREC;
  const int *cl = crs_ei,   *cr = crs_ei + ECROSS;
  float* out = (float*)d_out;

  char* base = (char*)d_ws;
  size_t off = 0;
  auto alloc = [&](size_t bytes) -> void* {
    void* r = base + off;
    off = (off + bytes + 255) & ~(size_t)255;
    return r;
  };
  float* emb     = (float*)alloc(NB * 32 * 4);
  float* lig_h   = (float*)alloc((size_t)NLIG * HDIM * 4);
  float* rec_h   = (float*)alloc((size_t)NREC * HDIM * 4);
  float* acc1    = (float*)alloc((size_t)NLIG * HDIM * 4);
  float* acc2    = (float*)alloc((size_t)NLIG * HDIM * 4);
  float* acc3    = (float*)alloc((size_t)NREC * HDIM * 4);
  float* acc4    = (float*)alloc((size_t)NREC * HDIM * 4);
  float* cnt1    = (float*)alloc((size_t)NLIG * 4);
  float* cnt2    = (float*)alloc((size_t)NLIG * 4);
  float* cnt3    = (float*)alloc((size_t)NREC * 4);
  float* cnt4    = (float*)alloc((size_t)NREC * 4);
  float* dl      = (float*)alloc((size_t)ELIG * 4);
  float* dr      = (float*)alloc((size_t)EREC * 4);
  float* dc      = (float*)alloc((size_t)ECROSS * 4);
  float* shl     = (float*)alloc((size_t)ELIG * 4 * 4);
  float* shr     = (float*)alloc((size_t)EREC * 4 * 4);
  float* shc     = (float*)alloc((size_t)ECROSS * 4 * 4);
  float* lig_ea  = (float*)alloc((size_t)ELIG * 16 * 4);
  float* rec_ea  = (float*)alloc((size_t)EREC * 16 * 4);
  float* crs_ea  = (float*)alloc((size_t)ECROSS * 16 * 4);
  float* dcen    = (float*)alloc((size_t)NLIG * 4);
  float* shcen   = (float*)alloc((size_t)NLIG * 4 * 4);
  float* cen_mlp = (float*)alloc((size_t)NLIG * 16 * 4);
  float* centers = (float*)alloc(NB * 3 * 4);
  float* ccnt    = (float*)alloc(NB * 4);
  float* gp      = (float*)alloc(NB * 12 * 4);
  float* gcnt    = (float*)alloc(NB * 4);
  __bf16* Xb  = (__bf16*)alloc((size_t)ELIG * 64 * 2);
  __bf16* W1b = (__bf16*)alloc((size_t)6 * 512 * 2);    // shuffled W1 fragments
  __bf16* W2b = (__bf16*)alloc((size_t)52 * 512 * 2);   // shuffled W2 fragments
  float* Y    = (float*)alloc((size_t)ECROSS * 416 * 4);

  auto g256 = [](long n) { return (unsigned)((n + 255) / 256); };
  auto zero = [&](float* p, long n) { k_zero<<<g256(n), 256, 0, stream>>>(p, n); };

  // stage A ---------------------------------------------------------------
  k_sin_emb<<<NB, 16, 0, stream>>>(t_tr, emb);
  k_node_init<<<g256(NLIG), 256, 0, stream>>>(lig_x, lig_batch, emb, P(6), P(7), lig_h, NLIG);
  k_node_init<<<g256(NREC), 256, 0, stream>>>(rec_x, rec_batch, emb, P(8), P(9), rec_h, NREC);
  k_edge_geom<<<g256(ELIG), 256, 0, stream>>>(lig_pos, lig_pos, lsrc, ldst, dl, shl, ELIG);
  k_edge_geom<<<g256(EREC), 256, 0, stream>>>(rec_pos, rec_pos, rsrc, rdst, dr, shr, EREC);
  k_edge_geom<<<g256(ECROSS), 256, 0, stream>>>(lig_pos, rec_pos, cl, cr, dc, shc, ECROSS);
  k_edge_mlp<<<g256(ELIG), 256, 0, stream>>>(lig_ef, lsrc, lig_batch, emb, dl, 5.0f,
                                             P(10), P(11), P(12), P(13), lig_ea, ELIG, 68, 0);
  k_edge_mlp<<<g256(EREC), 256, 0, stream>>>(nullptr, rsrc, rec_batch, emb, dr, 30.0f,
                                             P(14), P(15), P(16), P(17), rec_ea, EREC, 64, 0);
  k_edge_mlp<<<g256(ECROSS), 256, 0, stream>>>(nullptr, cl, lig_batch, emb, dc, 250.0f,
                                               P(18), P(19), P(20), P(21), crs_ea, ECROSS, 64, 0);

  auto mlp_blocks = [](long E) { return (unsigned)(((E + 15) / 16 + 7) / 8); };
  auto pack = [&](const float* W, __bf16* Wb, int K, int N, int NKT, int NNT) {
    k_pack_w_shuf<<<g256((long)NKT * NNT * 512), 256, 0, stream>>>(W, Wb, K, N, NKT, NNT);
  };

  // ---- layer 0 (MODE 0 TP, wn = 320) ------------------------------------
  {
    const int ND = 28;
    k_build_X<<<g256(ELIG), 256, 0, stream>>>(lig_ea, lig_h, lsrc, lig_h, ldst, Xb, ELIG, 64);
    pack(P(26), W1b, 48, 48, 2, 3); pack(P(28), W2b, 48, 320, 2, 20);
    k_mlp_wmma<64, 48, 320><<<mlp_blocks(ELIG), 256, 0, stream>>>(Xb, ELIG, P(27), W1b, W2b, P(29), Y);
    zero(acc1, (long)NLIG * HDIM); zero(cnt1, NLIG);
    k_tp<0><<<g256(ELIG), 256, 0, stream>>>(Y, 320, lig_h, ldst, shl, lsrc, acc1, cnt1, ELIG, HDIM);
    k_build_X<<<g256(EREC), 256, 0, stream>>>(rec_ea, rec_h, rsrc, rec_h, rdst, Xb, EREC, 64);
    pack(P(34), W1b, 48, 48, 2, 3); pack(P(36), W2b, 48, 320, 2, 20);
    k_mlp_wmma<64, 48, 320><<<mlp_blocks(EREC), 256, 0, stream>>>(Xb, EREC, P(35), W1b, W2b, P(37), Y);
    zero(acc3, (long)NREC * HDIM); zero(cnt3, NREC);
    k_tp<0><<<g256(EREC), 256, 0, stream>>>(Y, 320, rec_h, rdst, shr, rsrc, acc3, cnt3, EREC, HDIM);
    k_build_X<<<g256(ECROSS), 256, 0, stream>>>(crs_ea, lig_h, cl, rec_h, cr, Xb, ECROSS, 64);
    pack(P(50), W1b, 48, 48, 2, 3); pack(P(52), W2b, 48, 320, 2, 20);
    k_mlp_wmma<64, 48, 320><<<mlp_blocks(ECROSS), 256, 0, stream>>>(Xb, ECROSS, P(51), W1b, W2b, P(53), Y);
    zero(acc2, (long)NLIG * HDIM); zero(cnt2, NLIG);
    k_tp<0><<<g256(ECROSS), 256, 0, stream>>>(Y, 320, rec_h, cr, shc, cl, acc2, cnt2, ECROSS, HDIM);
    pack(P(42), W1b, 48, 48, 2, 3); pack(P(44), W2b, 48, 320, 2, 20);
    k_mlp_wmma<64, 48, 320><<<mlp_blocks(ECROSS), 256, 0, stream>>>(Xb, ECROSS, P(43), W1b, W2b, P(45), Y);
    zero(acc4, (long)NREC * HDIM); zero(cnt4, NREC);
    k_tp<0><<<g256(ECROSS), 256, 0, stream>>>(Y, 320, lig_h, cl, shc, cr, acc4, cnt4, ECROSS, HDIM);
    k_node_update<<<g256(NLIG), 256, 0, stream>>>(lig_h, acc1, cnt1, acc2, cnt2, NLIG, ND);
    k_node_update<<<g256(NREC), 256, 0, stream>>>(rec_h, acc3, cnt3, acc4, cnt4, NREC, ND);
  }

  // ---- layer 1 (MODE 1 TP, wn = 416; lig only) ---------------------------
  {
    const int ND = 40;
    k_build_X<<<g256(ELIG), 256, 0, stream>>>(lig_ea, lig_h, lsrc, lig_h, ldst, Xb, ELIG, 64);
    pack(P(30), W1b, 48, 48, 2, 3); pack(P(32), W2b, 48, 416, 2, 26);
    k_mlp_wmma<64, 48, 416><<<mlp_blocks(ELIG), 256, 0, stream>>>(Xb, ELIG, P(31), W1b, W2b, P(33), Y);
    zero(acc1, (long)NLIG * HDIM); zero(cnt1, NLIG);
    k_tp<1><<<g256(ELIG), 256, 0, stream>>>(Y, 416, lig_h, ldst, shl, lsrc, acc1, cnt1, ELIG, HDIM);
    k_build_X<<<g256(ECROSS), 256, 0, stream>>>(crs_ea, lig_h, cl, rec_h, cr, Xb, ECROSS, 64);
    pack(P(54), W1b, 48, 48, 2, 3); pack(P(56), W2b, 48, 416, 2, 26);
    k_mlp_wmma<64, 48, 416><<<mlp_blocks(ECROSS), 256, 0, stream>>>(Xb, ECROSS, P(55), W1b, W2b, P(57), Y);
    zero(acc2, (long)NLIG * HDIM); zero(cnt2, NLIG);
    k_tp<1><<<g256(ECROSS), 256, 0, stream>>>(Y, 416, rec_h, cr, shc, cl, acc2, cnt2, ECROSS, HDIM);
    k_node_update<<<g256(NLIG), 256, 0, stream>>>(lig_h, acc1, cnt1, acc2, cnt2, NLIG, ND);
  }

  // ---- final pseudo-conv to graph outputs --------------------------------
  zero(centers, NB * 3); zero(ccnt, NB);
  k_center_acc<<<g256(NLIG), 256, 0, stream>>>(lig_pos, lig_batch, centers, ccnt, NLIG);
  k_center_div<<<1, NB, 0, stream>>>(centers, ccnt);
  k_cen_geom<<<g256(NLIG), 256, 0, stream>>>(lig_pos, lig_batch, centers, dcen, shcen, NLIG);
  k_edge_mlp<<<g256(NLIG), 256, 0, stream>>>(nullptr, nullptr, lig_batch, emb, dcen, 30.0f,
                                             P(22), P(23), P(24), P(25), cen_mlp, NLIG, 64, 1);
  k_build_X<<<g256(NLIG), 256, 0, stream>>>(cen_mlp, lig_h, nullptr, nullptr, nullptr, Xb, NLIG, 32);
  pack(P(58), W1b, 32, 32, 1, 2); pack(P(60), W2b, 32, 64, 1, 4);
  k_mlp_wmma<32, 32, 64><<<mlp_blocks(NLIG), 256, 0, stream>>>(Xb, NLIG, P(59), W1b, W2b, P(61), Y);
  zero(gp, NB * 12); zero(gcnt, NB);
  k_tp<2><<<g256(NLIG), 256, 0, stream>>>(Y, 64, lig_h, nullptr, shcen, lig_batch, gp, gcnt, NLIG, 12);
  k_final<<<1, NB, 0, stream>>>(gp, gcnt, emb, P(62), P(63), P(64), P(65),
                                P(66), P(67), P(68), P(69), out);
}
